// Synthesis_prior_net_38972533244018
// MI455X (gfx1250) — matrix operand order repaired and checked by reading, not compiled, not run on hardware
//
#include <hip/hip_runtime.h>
#include <hip/hip_bf16.h>

typedef __attribute__((ext_vector_type(16))) _Float16 v16h;
typedef __attribute__((ext_vector_type(8)))  float    v8f;

#define CIN 192   // in-channels for every stage

// ---------------- workspace layout (bytes, all 256-aligned) ----------------
constexpr size_t SZ_W1P = 25ull * 192 * 192 * 2;   // folded stage1 weights f16 [tap][oc][ic]
constexpr size_t SZ_W2P = 25ull * 192 * 192 * 2;
constexpr size_t SZ_W3P = 9ull  * 320 * 192 * 2;
constexpr size_t SZ_B1  = 768;                     // 192 f32 (padded)
constexpr size_t SZ_B2  = 768;
constexpr size_t SZ_B3  = 1280;                    // 320 f32
constexpr size_t SZ_XH  = 8ull * 32 * 32 * 192 * 2;     // x NHWC f16
constexpr size_t SZ_Y1  = 8ull * 64 * 64 * 192 * 2;     // stage1 out NHWC f16
constexpr size_t SZ_Y2  = 8ull * 128 * 128 * 192 * 2;   // stage2 out NHWC f16

constexpr size_t OFF_W1P = 0;
constexpr size_t OFF_W2P = OFF_W1P + SZ_W1P;
constexpr size_t OFF_W3P = OFF_W2P + SZ_W2P;
constexpr size_t OFF_B1  = OFF_W3P + SZ_W3P;
constexpr size_t OFF_B2  = OFF_B1  + SZ_B1;
constexpr size_t OFF_B3  = OFF_B2  + SZ_B2;
constexpr size_t OFF_XH  = OFF_B3  + SZ_B3;
constexpr size_t OFF_Y1  = OFF_XH  + SZ_XH;
constexpr size_t OFF_Y2  = OFF_Y1  + SZ_Y1;

// ---------------- helpers ----------------
__device__ __forceinline__ v16h ld16(const _Float16* p0, const _Float16* p1) {
  union { uint4 q[2]; v16h h; } r;
  r.q[0] = *(const uint4*)p0;
  r.q[1] = *(const uint4*)p1;
  return r.h;
}

__device__ __forceinline__ v8f wmma_f16(v16h a, v16h b, v8f c) {
  return __builtin_amdgcn_wmma_f32_16x16x32_f16(false, a, false, b, (short)0, c,
                                                false, false);
}

// Fold 5 weight splits, transpose [i][ic][oc][ky][kx] -> f16 [tap][oc][ic]
__global__ void fold_w(const float* __restrict__ w, _Float16* __restrict__ wp,
                       int Cout, int KK) {
  int idx = blockIdx.x * blockDim.x + threadIdx.x;
  int total = KK * Cout * CIN;
  if (idx >= total) return;
  int ic = idx % CIN; int r = idx / CIN;
  int oc = r % Cout;  int t = r / Cout;
  size_t stride = (size_t)CIN * Cout * KK;
  size_t base   = ((size_t)ic * Cout + oc) * KK + t;
  float s = 0.f;
#pragma unroll
  for (int i = 0; i < 5; ++i) s += w[base + i * stride];
  wp[idx] = (_Float16)s;
}

__global__ void fold_b(const float* __restrict__ b, float* __restrict__ bs, int C) {
  int i = blockIdx.x * blockDim.x + threadIdx.x;
  if (i >= C) return;
  float s = 0.f;
#pragma unroll
  for (int j = 0; j < 5; ++j) s += b[(size_t)j * C + i];
  bs[i] = s;
}

// NCHW f32 -> NHWC f16
__global__ void to_nhwc(const float* __restrict__ x, _Float16* __restrict__ xh,
                        int Bn, int H, int W) {
  int idx = blockIdx.x * blockDim.x + threadIdx.x;
  int total = Bn * H * W * CIN;
  if (idx >= total) return;
  int ic = idx % CIN; int r = idx / CIN;
  int ix = r % W; r /= W;
  int iy = r % H; int img = r / H;
  xh[idx] = (_Float16)x[(((size_t)img * CIN + ic) * H + iy) * W + ix];
}

// ---------------- implicit-GEMM transposed conv via WMMA ----------------
// One wave computes a 64(out-ch) x 32(pixel) tile: 4x2 WMMA accumulators.
// Each A (weight) fragment feeds two B (pixel) fragments and each B fragment
// feeds four A fragments: 12 b128 loads per 8 WMMAs (1.5 loads/wmma).
// For S==2 all 32 pixels share a parity so the tap set is wave-uniform.
template<int Cout, int Hi, int Wi, int K, int S, int P, int MODE>
__global__ __launch_bounds__(128)
void deconv_wmma(const _Float16* __restrict__ X,   // NHWC f16 input
                 const _Float16* __restrict__ Wp,  // [tap][oc][ic] f16
                 const float* __restrict__ bias,   // [Cout]
                 const float* __restrict__ mul,    // [Cout]
                 const float* __restrict__ clp, const float* __restrict__ scl,
                 float add1, float rcp1, float add2, float rcp2,
                 _Float16* __restrict__ Yh, float* __restrict__ Yf) {
  constexpr int Bn = 8;
  constexpr int Ho = (S == 2) ? Hi * 2 : Hi;
  constexpr int Wo = (S == 2) ? Wi * 2 : Wi;
  constexpr int MG = Cout / 64;     // 64-out-channel groups
  constexpr int TXn = Wo / 32;      // 32-pixel column groups per row
  constexpr int total = Bn * Ho * TXn * MG;
  constexpr int sh = S - 1;         // 0 (stride 1) or 1 (stride 2)

  const int wid = blockIdx.x * (blockDim.x >> 5) + (threadIdx.x >> 5);
  if (wid >= total) return;

  int mg = wid % MG; int r0 = wid / MG;
  int txn = r0 % TXn; r0 /= TXn;
  int oy = r0 % Ho;   int img = r0 / Ho;
  // 32 same-parity columns: S==2 -> ox = p + 64*g + 2j ; S==1 -> ox = 32*g + j
  const int oxc = (S == 2) ? ((txn & 1) + ((txn >> 1) << 6)) : (txn << 5);

  const int lane = threadIdx.x & 31;
  const int nl = lane & 15;        // N index (pixel) / A row (oc)
  const int hl = lane >> 4;        // lane-half selector for K striping

  v8f acc[4][2] = {};
  const v16h bz = {};
  const _Float16* Abase = Wp + ((size_t)(mg * 64 + nl)) * CIN + hl * 16;

#pragma unroll 1
  for (int ky = 0; ky < K; ++ky) {
    int qy = oy + P - ky;
    if (qy < 0 || (qy & sh)) continue;
    int iy = qy >> sh;
    if (iy >= Hi) continue;
#pragma unroll 1
    for (int kx = 0; kx < K; ++kx) {
      int qx = oxc + P - kx;
      if (qx & sh) continue;                 // parity is wave-uniform
      int ix0 = (qx >> sh) + nl;             // per-lane input column, tile 0
      int ix1 = ix0 + 16;                    // tile 1
      bool bv0 = (ix0 >= 0) && (ix0 < Wi);
      bool bv1 = (ix1 >= 0) && (ix1 < Wi);
      const _Float16* Ap = Abase + (size_t)(ky * K + kx) * Cout * CIN;
      const _Float16* Bp0 =
          X + ((size_t)(img * Hi + iy) * Wi + ix0) * CIN + hl * 16;
      const _Float16* Bp1 = Bp0 + (size_t)16 * CIN;
#pragma unroll
      for (int c = 0; c < CIN / 32; ++c) {
        v16h b0, b1;
        if (bv0) b0 = ld16(Bp0 + c * 32, Bp0 + c * 32 + 8); else b0 = bz;
        if (bv1) b1 = ld16(Bp1 + c * 32, Bp1 + c * 32 + 8); else b1 = bz;
#pragma unroll
        for (int sub = 0; sub < 4; ++sub) {
          v16h a = ld16(Ap + sub * 16 * CIN + c * 32,
                        Ap + sub * 16 * CIN + c * 32 + 8);
          acc[sub][0] = wmma_f16(a, b0, acc[sub][0]);
          acc[sub][1] = wmma_f16(a, b1, acc[sub][1]);
        }
      }
    }
  }

  // fused quantization epilogue
  float clipv = 0.f, sclv = 0.f;
  if (MODE == 0) { clipv = clp[0]; sclv = scl[0]; }
  else if (MODE == 1) { clipv = clp[1]; sclv = scl[1]; }

#pragma unroll
  for (int sub = 0; sub < 4; ++sub) {
#pragma unroll
    for (int r = 0; r < 8; ++r) {
      int oc = mg * 64 + sub * 16 + r + hl * 8; // C/D layout: lanes 16-31 = M+8
      float bb = bias[oc];
      float mm = mul[oc];
#pragma unroll
      for (int n2 = 0; n2 < 2; ++n2) {
        int ox = oxc + S * (n2 * 16 + nl);
        float v = acc[sub][n2][r] + bb;
        v *= mm;
        v = floorf((v + add1) * rcp1);
        if (MODE < 2) {
          v = fminf(fmaxf(v, 0.f), clipv);
          v = floorf((v * sclv + add2) * rcp2);
          Yh[((size_t)(img * Ho + oy) * Wo + ox) * Cout + oc] = (_Float16)v;
        } else {
          Yf[(((size_t)img * Cout + oc) * Ho + oy) * Wo + ox] = v;
        }
      }
    }
  }
}

// ---------------- host ----------------
extern "C" void kernel_launch(void* const* d_in, const int* in_sizes, int n_in,
                              void* d_out, int out_size, void* d_ws, size_t ws_size,
                              hipStream_t stream) {
  const float* x    = (const float*)d_in[0];
  const float* w1   = (const float*)d_in[1];
  const float* b1   = (const float*)d_in[2];
  const float* w2   = (const float*)d_in[3];
  const float* b2   = (const float*)d_in[4];
  const float* w3   = (const float*)d_in[5];
  const float* b3   = (const float*)d_in[6];
  const float* mul0 = (const float*)d_in[7];
  const float* mul1 = (const float*)d_in[8];
  const float* mul2 = (const float*)d_in[9];
  const float* clp  = (const float*)d_in[10];
  const float* scl  = (const float*)d_in[11];

  char* ws = (char*)d_ws;
  _Float16* w1p = (_Float16*)(ws + OFF_W1P);
  _Float16* w2p = (_Float16*)(ws + OFF_W2P);
  _Float16* w3p = (_Float16*)(ws + OFF_W3P);
  float*    b1s = (float*)(ws + OFF_B1);
  float*    b2s = (float*)(ws + OFF_B2);
  float*    b3s = (float*)(ws + OFF_B3);
  _Float16* xh  = (_Float16*)(ws + OFF_XH);
  _Float16* y1  = (_Float16*)(ws + OFF_Y1);
  _Float16* y2  = (_Float16*)(ws + OFF_Y2);

  // fold the 5 redundant weight splits (linearity of deconv)
  {
    int t = 25 * 192 * 192;
    fold_w<<<(t + 255) / 256, 256, 0, stream>>>(w1, w1p, 192, 25);
    fold_w<<<(t + 255) / 256, 256, 0, stream>>>(w2, w2p, 192, 25);
    int t3 = 9 * 320 * 192;
    fold_w<<<(t3 + 255) / 256, 256, 0, stream>>>(w3, w3p, 320, 9);
    fold_b<<<1, 256, 0, stream>>>(b1, b1s, 192);
    fold_b<<<1, 256, 0, stream>>>(b2, b2s, 192);
    fold_b<<<2, 256, 0, stream>>>(b3, b3s, 320);
  }
  // input NCHW f32 -> NHWC f16 (values are small integers: exact in f16)
  {
    int t = 8 * 32 * 32 * 192;
    to_nhwc<<<(t + 255) / 256, 256, 0, stream>>>(x, xh, 8, 32, 32);
  }

  // stage 1: 192->192, 5x5, stride 2, pad 2, out_pad 1 : 32x32 -> 64x64
  {
    constexpr int tiles = 8 * 64 * (64 / 32) * (192 / 64);
    deconv_wmma<192, 32, 32, 5, 2, 2, 0><<<(tiles + 3) / 4, 128, 0, stream>>>(
        xh, w1p, b1s, mul0, clp, scl,
        256.f, 1.f / 512.f, 1048576.f, 1.f / 2097152.f, y1, nullptr);
  }
  // stage 2: 192->192, 5x5, stride 2, pad 2, out_pad 1 : 64x64 -> 128x128
  {
    constexpr int tiles = 8 * 128 * (128 / 32) * (192 / 64);
    deconv_wmma<192, 64, 64, 5, 2, 2, 1><<<(tiles + 3) / 4, 128, 0, stream>>>(
        y1, w2p, b2s, mul1, clp, scl,
        512.f, 1.f / 1024.f, 1048576.f, 1.f / 2097152.f, y2, nullptr);
  }
  // stage 3: 192->320, 3x3, stride 1, pad 1 : 128x128 -> 128x128, f32 NCHW out
  {
    constexpr int tiles = 8 * 128 * (128 / 32) * (320 / 64);
    deconv_wmma<320, 128, 128, 3, 1, 1, 2><<<(tiles + 3) / 4, 128, 0, stream>>>(
        y2, w3p, b3s, mul2, clp, scl,
        8192.f, 1.f / 16384.f, 0.f, 1.f, nullptr, (float*)d_out);
  }
}
